// CaptionAttentionNet_50379966382120
// MI455X (gfx1250) — matrix operand order repaired and checked by reading, not compiled, not run on hardware
//
#include <hip/hip_runtime.h>

// ---------------------------------------------------------------------------
// CDNA5 (gfx1250) CaptionAttentionNet forward.
//  * all matmuls via v_wmma_f32_16x16x32_bf16 (bf16 in, fp32 accumulate)
//  * concats folded into accumulating GEMMs over weight column slices
//  * double-buffered LDS staging; global->LDS copies use the gfx1250 async
//    path (GLOBAL_LOAD_ASYNC_TO_LDS_B128 + s_wait_asynccnt) when available
//  * staging guards are clamp-based (no exec-mask divergence); only the
//    epilogue store is predicated (logits N=10000 edge tile)
// ---------------------------------------------------------------------------

typedef __attribute__((ext_vector_type(16))) __bf16 v16bf;
typedef __attribute__((ext_vector_type(8)))  float  v8f;
typedef __attribute__((ext_vector_type(4)))  int    v4i;

#define TILE_M 128
#define TILE_N 128
#define TILE_K 32
#define LDS_STRIDE 40   // 80B row pitch -> conflict-free b128 fragment loads

#if __has_builtin(__builtin_amdgcn_global_load_async_to_lds_b128) && \
    __has_builtin(__builtin_amdgcn_s_wait_asynccnt)
#define USE_ASYNC 1
#else
#define USE_ASYNC 0
#endif

typedef __attribute__((address_space(1))) v4i as1_v4i;   // global int4
typedef __attribute__((address_space(3))) v4i as3_v4i;   // LDS int4

__device__ __forceinline__ void copy_b128(const __bf16* g, __bf16* l)
{
#if USE_ASYNC
    // generic->AS1 is bit-identical; generic LDS pointer's low 32 bits are the
    // wave-relative LDS byte offset (aperture lives in the high bits).
    as1_v4i* gp = (as1_v4i*)(unsigned long long)(uintptr_t)g;
    as3_v4i* lp = (as3_v4i*)(unsigned int)(uintptr_t)l;
    __builtin_amdgcn_global_load_async_to_lds_b128(gp, lp, 0, 0);
#else
    *(uint4*)l = *(const uint4*)g;
#endif
}

#if USE_ASYNC
#define WAIT_ASYNC(n) __builtin_amdgcn_s_wait_asynccnt(n)
#else
#define WAIT_ASYNC(n) ((void)0)
#endif

union Frag { v16bf v; uint4 u[2]; };

// C[M,N] = A[M,K] * B[N,K]^T  (+ bias[n]) (+ Cin[m,n]); writes f32 and/or bf16.
__global__ __launch_bounds__(256, 2) void gemm_bf16_wmma(
    const __bf16* __restrict__ A, int lda,
    const __bf16* __restrict__ B, int ldb,
    const float*  __restrict__ bias,
    const float*  __restrict__ Cin, long ldcin,
    float*        __restrict__ Cout,
    __bf16*       __restrict__ Cbf,
    long ldc, int M, int N, int K)
{
    __shared__ __bf16 As[2][TILE_M][LDS_STRIDE];
    __shared__ __bf16 Bs[2][TILE_N][LDS_STRIDE];

    const int tid  = threadIdx.x;
    const int lane = tid & 31;
    const int wv   = tid >> 5;   // 8 waves
    const int wm   = wv >> 2;    // 0..1 : 64-row slab
    const int wn   = wv & 3;     // 0..3 : 32-col slab
    const int m0   = blockIdx.y * TILE_M;
    const int n0   = blockIdx.x * TILE_N;

    // ---- clamped per-thread staging addresses (4 x b128 per stage) ----
    const int ra  = tid >> 2;          // 0..63
    const int col = (tid & 3) << 3;    // 0,8,16,24 (bf16 units)
    int gmA0 = m0 + ra;       if (gmA0 > M - 1) gmA0 = M - 1;
    int gmA1 = m0 + ra + 64;  if (gmA1 > M - 1) gmA1 = M - 1;
    int gnB0 = n0 + ra;       if (gnB0 > N - 1) gnB0 = N - 1;
    int gnB1 = n0 + ra + 64;  if (gnB1 > N - 1) gnB1 = N - 1;
    const __bf16* pA0 = A + (long)gmA0 * lda + col;
    const __bf16* pA1 = A + (long)gmA1 * lda + col;
    const __bf16* pB0 = B + (long)gnB0 * ldb + col;
    const __bf16* pB1 = B + (long)gnB1 * ldb + col;

    v8f zero = {};
    v8f acc[4][2];
#pragma unroll
    for (int r = 0; r < 4; ++r)
#pragma unroll
        for (int s = 0; s < 2; ++s)
            acc[r][s] = zero;

    const int fr = lane & 15;   // row within 16x16 fragment
    const int hi = lane >> 4;   // lane-group selector

    auto issue = [&](int buf, int k0) {
        copy_b128(pA0 + k0, &As[buf][ra][col]);
        copy_b128(pA1 + k0, &As[buf][ra + 64][col]);
        copy_b128(pB0 + k0, &Bs[buf][ra][col]);
        copy_b128(pB1 + k0, &Bs[buf][ra + 64][col]);
    };

    issue(0, 0);
    int buf = 0;
    for (int k0 = 0; k0 < K; k0 += TILE_K) {
        const bool have_next = (k0 + TILE_K) < K;
        if (have_next) {
            issue(buf ^ 1, k0 + TILE_K);   // DMA next stage during this compute
            WAIT_ASYNC(4);                 // first 4 (this stage) have landed
        } else {
            WAIT_ASYNC(0);
        }
        __syncthreads();                   // stage data visible to all waves

        // A fragment: lane group 0 holds K{0-7,16-23}, group 1 K{8-15,24-31}.
        Frag af[4];
#pragma unroll
        for (int r = 0; r < 4; ++r) {
            const __bf16* p = &As[buf][wm * 64 + r * 16 + fr][hi * 8];
            af[r].u[0] = *(const uint4*)(p);
            af[r].u[1] = *(const uint4*)(p + 16);
        }
        // B fragment: lane group 0 holds K{0-7,16-23}, group 1 K{8-15,24-31}.
        Frag bfr[2];
#pragma unroll
        for (int s = 0; s < 2; ++s) {
            const __bf16* p = &Bs[buf][wn * 32 + s * 16 + fr][hi * 8];
            bfr[s].u[0] = *(const uint4*)(p);
            bfr[s].u[1] = *(const uint4*)(p + 16);
        }
#pragma unroll
        for (int r = 0; r < 4; ++r)
#pragma unroll
            for (int s = 0; s < 2; ++s)
                acc[r][s] = __builtin_amdgcn_wmma_f32_16x16x32_bf16(
                    false, af[r].v, false, bfr[s].v, (short)0, acc[r][s],
                    false, false);

        __syncthreads();                   // all reads of buf done before rewrite
        buf ^= 1;
    }

    // Epilogue: lane holds col n = lane%16; VGPR i holds row m = base + i,
    // base offset 8 for lanes 16..31.
#pragma unroll
    for (int r = 0; r < 4; ++r) {
#pragma unroll
        for (int s = 0; s < 2; ++s) {
            int n  = n0 + wn * 32 + s * 16 + fr;
            int mb = m0 + wm * 64 + r * 16 + (hi << 3);
            if (n < N) {
                float bv = bias ? bias[n] : 0.0f;
#pragma unroll
                for (int i = 0; i < 8; ++i) {
                    int m = mb + i;
                    if (m < M) {
                        float v = acc[r][s][i] + bv;
                        if (Cin)  v += Cin[(long)m * ldcin + n];
                        if (Cout) Cout[(long)m * ldc + n] = v;
                        if (Cbf)  Cbf[(long)m * ldc + n] = (__bf16)v;
                    }
                }
            }
        }
    }
}

__global__ void cvt_f32_bf16(const float* __restrict__ s, __bf16* __restrict__ d, long n)
{
    long i = (long)blockIdx.x * blockDim.x + threadIdx.x;
    if (i < n) d[i] = (__bf16)s[i];
}

__global__ void gather_emb(const float* __restrict__ emb, const int* __restrict__ ix,
                           __bf16* __restrict__ x, int E)
{
    int row = blockIdx.x;                 // b*T + t
    int id  = ix[row];
    const float* src = emb + (long)id * E;
    __bf16* dst = x + (long)row * E;
    for (int e = threadIdx.x; e < E; e += blockDim.x)
        dst[e] = (__bf16)src[e];
}

// gates[b, 0:4D] laid out (i,f,g,o); c updated in place; h written as bf16
// into the per-step buffer and into the [B,T,D] history. D hardcoded = 512.
__global__ void lstm_cell(const float* __restrict__ gates, float* __restrict__ c,
                          __bf16* __restrict__ hbf, __bf16* __restrict__ hs,
                          int t, int T)
{
    const int D = 512;
    int i = blockIdx.x * blockDim.x + threadIdx.x;   // over B*D (exact grid)
    int b = i >> 9, d = i & (D - 1);
    const float* g = gates + ((long)b << 11);
    float ig = 1.0f / (1.0f + __expf(-g[d]));
    float fg = 1.0f / (1.0f + __expf(-g[D + d]));
    float gg = tanhf(g[2 * D + d]);
    float og = 1.0f / (1.0f + __expf(-g[3 * D + d]));
    float cv = fg * c[i] + ig * gg;
    c[i] = cv;
    float hv = og * tanhf(cv);
    hbf[i] = (__bf16)hv;
    hs[((long)b * T + t) * D + d] = (__bf16)hv;
}

extern "C" void kernel_launch(void* const* d_in, const int* in_sizes, int n_in,
                              void* d_out, int out_size, void* d_ws, size_t ws_size,
                              hipStream_t stream)
{
    (void)in_sizes; (void)n_in; (void)out_size; (void)ws_size;
    constexpr int Bz = 128, T = 64, Vv = 10000, E = 512, D = 512, F = 2048;
    constexpr int BT = Bz * T;            // 8192

    const float* img   = (const float*)d_in[0];
    const int*   ix    = (const int*)  d_in[1];
    const float* emb   = (const float*)d_in[2];
    const float* Wih[4]  = { (const float*)d_in[3], (const float*)d_in[5],
                             (const float*)d_in[7], (const float*)d_in[9] };   // init h1,c1,h2,c2
    const float* bih[4]  = { (const float*)d_in[4], (const float*)d_in[6],
                             (const float*)d_in[8], (const float*)d_in[10] };
    const float* Wih1  = (const float*)d_in[11];
    const float* Whh1  = (const float*)d_in[12];
    const float* b1    = (const float*)d_in[13];
    const float* Wih2  = (const float*)d_in[14];
    const float* Whh2  = (const float*)d_in[15];
    const float* b2    = (const float*)d_in[16];
    const float* Wv1   = (const float*)d_in[17];
    const float* bv1   = (const float*)d_in[18];
    const float* Wo1   = (const float*)d_in[19];
    const float* bo1   = (const float*)d_in[20];
    const float* Wv2   = (const float*)d_in[21];
    const float* bv2   = (const float*)d_in[22];
    const float* Wo2   = (const float*)d_in[23];
    const float* bo2   = (const float*)d_in[24];
    const float* WL    = (const float*)d_in[25];
    const float* bL    = (const float*)d_in[26];
    float* out = (float*)d_out;           // [B,T,V] fp32

    // ---- bump allocator on d_ws ----
    uintptr_t cur = (uintptr_t)d_ws;
    auto abf = [&](size_t n) { __bf16* p = (__bf16*)cur; cur += ((n * 2 + 255) & ~(size_t)255); return p; };
    auto af32 = [&](size_t n) { float*  p = (float*) cur; cur += ((n * 4 + 255) & ~(size_t)255); return p; };

    __bf16* bWinit[4]; for (int i = 0; i < 4; ++i) bWinit[i] = abf((size_t)D * F);
    __bf16* bWih1 = abf((size_t)4 * D * E);
    __bf16* bWhh1 = abf((size_t)4 * D * D);
    __bf16* bWih2 = abf((size_t)4 * D * 2 * D);
    __bf16* bWhh2 = abf((size_t)4 * D * D);
    __bf16* bWv1  = abf((size_t)D * D);
    __bf16* bWo1  = abf((size_t)D * D);
    __bf16* bWv2  = abf((size_t)D * D);
    __bf16* bWo2  = abf((size_t)D * D);
    __bf16* bWL   = abf((size_t)Vv * 3 * D);
    __bf16* bimg  = abf((size_t)Bz * F);
    __bf16* bx    = abf((size_t)BT * E);
    __bf16* bh1s  = abf((size_t)BT * D);
    __bf16* batt1 = abf((size_t)BT * D);
    __bf16* bh2s  = abf((size_t)BT * D);
    __bf16* batt2 = abf((size_t)BT * D);
    __bf16* bvtmp = abf((size_t)BT * D);
    __bf16* bh1   = abf((size_t)Bz * D);
    __bf16* bh2   = abf((size_t)Bz * D);
    float* c1f    = af32((size_t)Bz * D);
    float* c2f    = af32((size_t)Bz * D);
    float* gates  = af32((size_t)Bz * 4 * D);
    float* xp     = af32((size_t)BT * 4 * D);   // reused for layer1 then layer2

    auto cvt = [&](const float* s, __bf16* dst, long n) {
        cvt_f32_bf16<<<dim3((unsigned)((n + 255) / 256)), dim3(256), 0, stream>>>(s, dst, n);
    };
    auto gemm = [&](const __bf16* A, int lda, const __bf16* Bm, int ldb,
                    const float* bias, const float* Cin, long ldcin,
                    float* Cout, __bf16* Cbf, long ldc, int M, int N, int K) {
        dim3 g((unsigned)((N + TILE_N - 1) / TILE_N), (unsigned)((M + TILE_M - 1) / TILE_M));
        gemm_bf16_wmma<<<g, dim3(256), 0, stream>>>(A, lda, Bm, ldb, bias, Cin, ldcin,
                                                    Cout, Cbf, ldc, M, N, K);
    };

    // ---- weight / input conversion to bf16 ----
    for (int i = 0; i < 4; ++i) cvt(Wih[i], bWinit[i], (long)D * F);
    cvt(Wih1, bWih1, (long)4 * D * E);
    cvt(Whh1, bWhh1, (long)4 * D * D);
    cvt(Wih2, bWih2, (long)4 * D * 2 * D);
    cvt(Whh2, bWhh2, (long)4 * D * D);
    cvt(Wv1, bWv1, (long)D * D);  cvt(Wo1, bWo1, (long)D * D);
    cvt(Wv2, bWv2, (long)D * D);  cvt(Wo2, bWo2, (long)D * D);
    cvt(WL,  bWL,  (long)Vv * 3 * D);
    cvt(img, bimg, (long)Bz * F);
    gather_emb<<<dim3(BT), dim3(256), 0, stream>>>(emb, ix, bx, E);

    // ---- initial states: h/c = img @ W_init^T + b ----
    gemm(bimg, F, bWinit[0], F, bih[0], nullptr, 0, nullptr, bh1, D, Bz, D, F); // h1 (bf16)
    gemm(bimg, F, bWinit[1], F, bih[1], nullptr, 0, c1f, nullptr, D, Bz, D, F); // c1 (f32)
    gemm(bimg, F, bWinit[2], F, bih[2], nullptr, 0, nullptr, bh2, D, Bz, D, F); // h2
    gemm(bimg, F, bWinit[3], F, bih[3], nullptr, 0, c2f, nullptr, D, Bz, D, F); // c2

    // ---- layer 1: xp1 = x @ W_ih1^T + b1, then 64-step scan ----
    gemm(bx, E, bWih1, E, b1, nullptr, 0, xp, nullptr, 4 * D, BT, 4 * D, E);
    for (int t = 0; t < T; ++t) {
        gemm(bh1, D, bWhh1, D, nullptr, xp + (long)t * 4 * D, (long)T * 4 * D,
             gates, nullptr, 4 * D, Bz, 4 * D, D);
        lstm_cell<<<dim3((Bz * D) / 256), dim3(256), 0, stream>>>(gates, c1f, bh1, bh1s, t, T);
    }

    // ---- attn1 (length-1 key => two linear layers) ----
    gemm(bh1s,  D, bWv1, D, bv1, nullptr, 0, nullptr, bvtmp, D, BT, D, D);
    gemm(bvtmp, D, bWo1, D, bo1, nullptr, 0, nullptr, batt1, D, BT, D, D);

    // ---- layer 2: xp2 = [h1s, attn1] @ W_ih2^T + b2 (split-K accumulate) ----
    gemm(bh1s,  D, bWih2,     2 * D, b2,      nullptr, 0,        xp, nullptr, 4 * D, BT, 4 * D, D);
    gemm(batt1, D, bWih2 + D, 2 * D, nullptr, xp, 4 * D,         xp, nullptr, 4 * D, BT, 4 * D, D);
    for (int t = 0; t < T; ++t) {
        gemm(bh2, D, bWhh2, D, nullptr, xp + (long)t * 4 * D, (long)T * 4 * D,
             gates, nullptr, 4 * D, Bz, 4 * D, D);
        lstm_cell<<<dim3((Bz * D) / 256), dim3(256), 0, stream>>>(gates, c2f, bh2, bh2s, t, T);
    }

    // ---- attn2 ----
    gemm(bh2s,  D, bWv2, D, bv2, nullptr, 0, nullptr, bvtmp, D, BT, D, D);
    gemm(bvtmp, D, bWo2, D, bo2, nullptr, 0, nullptr, batt2, D, BT, D, D);

    // ---- logits: out = [h2s, attn1, attn2] @ W_logits^T + b (3 accumulating GEMMs) ----
    gemm(bh2s,  D, bWL,         3 * D, bL,      nullptr, 0,  out, nullptr, Vv, BT, Vv, D);
    gemm(batt1, D, bWL + D,     3 * D, nullptr, out, Vv,     out, nullptr, Vv, BT, Vv, D);
    gemm(batt2, D, bWL + 2 * D, 3 * D, nullptr, out, Vv,     out, nullptr, Vv, BT, Vv, D);
}
